// CNNModel_34411277976076
// MI455X (gfx1250) — compile-verified, matrix operand order checked by reading, not run on hardware
//
#include <hip/hip_runtime.h>
#include <math.h>

typedef __bf16 bf16;
typedef __attribute__((ext_vector_type(8)))  bf16  v8bf;
typedef __attribute__((ext_vector_type(16))) bf16  v16bf;
typedef __attribute__((ext_vector_type(8)))  float v8f;
typedef __attribute__((ext_vector_type(2)))  float v2f;

#define EPS 1e-5f

// ---------------------------------------------------------------------------
// helpers
// ---------------------------------------------------------------------------
__device__ __forceinline__ v16bf ldab(const bf16* p) {
  // CDNA5 16-bit A/B operand fragment: elems 0..7 = K[0..7] (this lane's K base),
  // elems 8..15 = K[16..23] relative to same base (see CDNA5 ISA layout tables).
  v8bf lo = *(const v8bf*)(p);
  v8bf hi = *(const v8bf*)(p + 16);
  v16bf r;
#pragma unroll
  for (int i = 0; i < 8; ++i) { r[i] = lo[i]; r[i + 8] = hi[i]; }
  return r;
}

// ---------------------------------------------------------------------------
// CNN front-end (bandwidth-trivial, plain VALU)
// ---------------------------------------------------------------------------
__global__ void k_conv1_pool(const float* __restrict__ x,      // (16,2048,64)
                             const float* __restrict__ w,      // (8,1,K,K)
                             const float* __restrict__ bias,   // (8)
                             const float* __restrict__ bg, const float* __restrict__ bb,
                             const float* __restrict__ bm, const float* __restrict__ bv,
                             float* __restrict__ out,          // (16,8,PH,PW)
                             int K, int S, int PH, int PW) {
  int idx = blockIdx.x * blockDim.x + threadIdx.x;
  int total = 16 * 8 * PH * PW;
  if (idx >= total) return;
  int pw = idx % PW; int t = idx / PW;
  int ph = t % PH; t /= PH;
  int c = t % 8; int b = t / 8;
  float sc = bg[c] / sqrtf(bv[c] + EPS);
  float sh = bb[c] - bm[c] * sc;
  const float* wc = w + c * K * K;
  float m = -3.4e38f;
  for (int py = 0; py < 2; ++py)
    for (int px = 0; px < 2; ++px) {
      int iy0 = (ph * 2 + py) * S, ix0 = (pw * 2 + px) * S;
      float acc = bias[c];
      for (int ky = 0; ky < K; ++ky) {
        const float* xr = x + (b * 2048 + iy0 + ky) * 64 + ix0;
        const float* wr = wc + ky * K;
        for (int kx = 0; kx < K; ++kx) acc += xr[kx] * wr[kx];
      }
      float v = fmaxf(acc * sc + sh, 0.f);
      m = fmaxf(m, v);
    }
  out[idx] = m;
}

__global__ void k_conv2(const float* __restrict__ in,   // (16,8,PH,PW)
                        const float* __restrict__ w,    // (32,8,K,K)
                        const float* __restrict__ bias, // (32)
                        const float* __restrict__ bg, const float* __restrict__ bb,
                        const float* __restrict__ bm, const float* __restrict__ bv,
                        float* __restrict__ out,        // (16,32,H2,W2)
                        int K, int S, int PH, int PW, int H2, int W2) {
  int idx = blockIdx.x * blockDim.x + threadIdx.x;
  int total = 16 * 32 * H2 * W2;
  if (idx >= total) return;
  int xw = idx % W2; int t = idx / W2;
  int y = t % H2; t /= H2;
  int c = t % 32; int b = t / 32;
  float sc = bg[c] / sqrtf(bv[c] + EPS);
  float sh = bb[c] - bm[c] * sc;
  float acc = bias[c];
  int iy0 = y * S, ix0 = xw * S;
  for (int ic = 0; ic < 8; ++ic) {
    const float* ip = in + ((b * 8 + ic) * PH + iy0) * PW + ix0;
    const float* wp = w + ((c * 8 + ic) * K) * K;
    for (int ky = 0; ky < K; ++ky)
      for (int kx = 0; kx < K; ++kx) acc += ip[ky * PW + kx] * wp[ky * K + kx];
  }
  out[idx] = fmaxf(acc * sc + sh, 0.f);
}

__global__ void k_conv3(const float* __restrict__ in,   // (16,32,H,W)
                        const float* __restrict__ w,    // (4,32)
                        const float* __restrict__ bias, // (4)
                        const float* __restrict__ bg, const float* __restrict__ bb,
                        const float* __restrict__ bm, const float* __restrict__ bv,
                        float* __restrict__ out,        // (16,4,H,W)
                        int H, int W) {
  int idx = blockIdx.x * blockDim.x + threadIdx.x;
  int hw = H * W;
  int total = 16 * 4 * hw;
  if (idx >= total) return;
  int pos = idx % hw; int t = idx / hw;
  int c = t % 4; int b = t / 4;
  float sc = bg[c] / sqrtf(bv[c] + EPS);
  float sh = bb[c] - bm[c] * sc;
  float acc = bias[c];
  const float* ip = in + (b * 32) * hw + pos;
  const float* wp = w + c * 32;
  for (int ic = 0; ic < 32; ++ic) acc += ip[ic * hw] * wp[ic];
  out[idx] = fmaxf(acc * sc + sh, 0.f);
}

__global__ void k_resize(const float* __restrict__ r3,  // (16,4,H,Wd)
                         const int* __restrict__ lengths,
                         float* __restrict__ feats,     // (16,10740)
                         int H, int Wd, int OH, int OW, int foff) {
  int idx = blockIdx.x * blockDim.x + threadIdx.x;
  int total = 16 * 4 * OH * OW;
  if (idx >= total) return;
  int xw = idx % OW; int t = idx / OW;
  int y = t % OH; t /= OH;
  int c = t % 4; int b = t / 4;
  int ml = 0;
  for (int i = 0; i < 16; ++i) ml = max(ml, lengths[i]);
  long long srch = (long long)((double)H * (double)lengths[b] / (double)ml);
  if (srch < 1) srch = 1;
  float sh = (float)srch;
  float iy = (y + 0.5f) * (sh / (float)OH) - 0.5f;
  iy = fminf(fmaxf(iy, 0.f), sh - 1.f);
  int y0 = (int)floorf(iy);
  int y1 = min(y0 + 1, (int)srch - 1);
  float wy = iy - (float)y0;
  float ix = (xw + 0.5f) * ((float)Wd / (float)OW) - 0.5f;
  ix = fminf(fmaxf(ix, 0.f), (float)Wd - 1.f);
  int x0 = (int)floorf(ix);
  int x1 = min(x0 + 1, Wd - 1);
  float wx = ix - (float)x0;
  const float* img = r3 + (b * 4 + c) * H * Wd;
  float v0 = img[y0 * Wd + x0] * (1.f - wy) + img[y1 * Wd + x0] * wy;
  float v1 = img[y0 * Wd + x1] * (1.f - wy) + img[y1 * Wd + x1] * wy;
  feats[b * 10740 + foff + (c * OH + y) * OW + xw] = v0 * (1.f - wx) + v1 * wx;
}

// ---------------------------------------------------------------------------
// conversion / layout kernels
// ---------------------------------------------------------------------------
__global__ void k_cvt_pad(const float* __restrict__ in, bf16* __restrict__ out,
                          int rows, int K, int Kp) {
  int idx = blockIdx.x * blockDim.x + threadIdx.x;
  if (idx >= rows * Kp) return;
  int r = idx / Kp, k = idx - r * Kp;
  out[idx] = (k < K) ? (bf16)in[r * K + k] : (bf16)0.f;
}

__global__ void k_build_A0(const float* __restrict__ batch,  // (16,2048,64)
                           const float* __restrict__ r,      // (16,1024)
                           bf16* __restrict__ A) {           // (16384,1152), row=t*16+b
  int idx = blockIdx.x * blockDim.x + threadIdx.x;
  if (idx >= 16384 * 1152) return;
  int d = idx % 1152; int row = idx / 1152;
  int b = row & 15, t = row >> 4;
  float v = (d < 128) ? batch[(b * 2048 + t * 2 + (d >> 6)) * 64 + (d & 63)]
                      : r[b * 1024 + (d - 128)];
  A[idx] = (bf16)v;
}

__global__ void k_build_A1(const float* __restrict__ h,  // (16,1024,512)
                           bf16* __restrict__ A) {       // (16384,512), row=t*16+b
  int idx = blockIdx.x * blockDim.x + threadIdx.x;
  if (idx >= 16384 * 512) return;
  int d = idx & 511; int row = idx >> 9;
  int b = row & 15, t = row >> 4;
  A[idx] = (bf16)h[((b << 10) + t) * 512 + d];
}

__global__ void k_scatter_back(const float* __restrict__ ob,   // (16,1024,256)
                               const int* __restrict__ lengths,
                               float* __restrict__ hl) {       // (16,1024,512) [...,256:]
  int idx = blockIdx.x * blockDim.x + threadIdx.x;
  if (idx >= 16 * 1024 * 256) return;
  int j = idx & 255; int t = (idx >> 8) & 1023; int b = idx >> 18;
  int nv = lengths[b] >> 1;
  int rt = nv - 1 - t;
  if (rt < 0) rt = 0;
  hl[((b << 10) + t) * 512 + 256 + j] = ob[((b << 10) + rt) * 256 + j];
}

// ---------------------------------------------------------------------------
// bf16 WMMA GEMM:  C(MxN) = A(M,Kp) * Bt(N,Kp)^T + bias   (f32 accumulate)
// one wave = one 16x64 output strip; emits v_wmma_f32_16x16x32_bf16
// ---------------------------------------------------------------------------
__global__ void __launch_bounds__(256) k_gemm_bf16(
    const bf16* __restrict__ A, const bf16* __restrict__ B,
    const float* __restrict__ bias, float* __restrict__ C,
    int M, int N, int Kp, int ngroups) {  // ngroups = N/64
  int wid = blockIdx.x * (blockDim.x >> 5) + (threadIdx.x >> 5);
  int totalw = (M >> 4) * ngroups;
  if (wid >= totalw) return;
  int mt = wid / ngroups, ng = wid - mt * ngroups;
  int lane = threadIdx.x & 31;
  int lr = lane & 15;
  int ko = (lane >> 4) * 8;
  const bf16* Ap = A + (size_t)(mt * 16 + lr) * Kp + ko;
  const bf16* B0 = B + (size_t)(ng * 64 + lr) * Kp + ko;
  const bf16* B1 = B0 + (size_t)16 * Kp;
  const bf16* B2 = B1 + (size_t)16 * Kp;
  const bf16* B3 = B2 + (size_t)16 * Kp;
  v8f c0 = {0,0,0,0,0,0,0,0}, c1 = c0, c2 = c0, c3 = c0;
  for (int k = 0; k < Kp; k += 32) {
    v16bf a = ldab(Ap + k);
    v16bf b0 = ldab(B0 + k);
    v16bf b1 = ldab(B1 + k);
    v16bf b2 = ldab(B2 + k);
    v16bf b3 = ldab(B3 + k);
    c0 = __builtin_amdgcn_wmma_f32_16x16x32_bf16(false, a, false, b0, (short)0, c0, false, false);
    c1 = __builtin_amdgcn_wmma_f32_16x16x32_bf16(false, a, false, b1, (short)0, c1, false, false);
    c2 = __builtin_amdgcn_wmma_f32_16x16x32_bf16(false, a, false, b2, (short)0, c2, false, false);
    c3 = __builtin_amdgcn_wmma_f32_16x16x32_bf16(false, a, false, b3, (short)0, c3, false, false);
  }
  int n0 = ng * 64 + lr;
  int mbase = mt * 16 + ((lane >> 4) << 3);
  float bi0 = bias[n0], bi1 = bias[n0 + 16], bi2 = bias[n0 + 32], bi3 = bias[n0 + 48];
#pragma unroll
  for (int r = 0; r < 8; ++r) {
    size_t mrow = (size_t)(mbase + r) * N;
    C[mrow + n0]      = c0[r] + bi0;
    C[mrow + n0 + 16] = c1[r] + bi1;
    C[mrow + n0 + 32] = c2[r] + bi2;
    C[mrow + n0 + 48] = c3[r] + bi3;
  }
}

// ---------------------------------------------------------------------------
// GRU recurrent scan. grid=(2): block 0 = forward, block 1 = backward.
// 16 waves; per step: hp(16x768) = h(16x256) @ whh^T + bhh via f32 WMMA
// (v_wmma_f32_16x16x4_f32, h resident in LDS, whh L2-resident), then gates.
// Backward reads xp rows at reversed time indices (projection precomputed
// unreversed); its raw scan outputs go to ob and are un-reversed afterwards.
// ---------------------------------------------------------------------------
__global__ void __launch_bounds__(512) k_gru_scan(
    const float* __restrict__ xp_f, const float* __restrict__ xp_b,
    const float* __restrict__ whh_f, const float* __restrict__ whh_b,
    const float* __restrict__ bhh_f, const float* __restrict__ bhh_b,
    const int* __restrict__ lengths,
    float* __restrict__ out_f,   // (16,1024,512), writes [...,0:256]
    float* __restrict__ out_b) { // (16,1024,256)
  __shared__ float h_lds[16 * 256];
  __shared__ float hp_lds[16 * 768];
  int dir = blockIdx.x;
  const float* xp  = dir ? xp_b  : xp_f;
  const float* whh = dir ? whh_b : whh_f;
  const float* bhh = dir ? bhh_b : bhh_f;
  float* out  = dir ? out_b : out_f;
  int out_ld  = dir ? 256 : 512;

  int tid = threadIdx.x;
  for (int i = tid; i < 16 * 256; i += 512) h_lds[i] = 0.f;
  __syncthreads();

  int lane = tid & 31, wave = tid >> 5;   // 16 waves
  int nt0 = wave * 3;                     // 3 N-tiles per wave (48 total)
  int lr = lane & 15;
  int kb = (lane >> 4) * 2;
  const float* Bp0 = whh + (size_t)(nt0 * 16 + lr) * 256 + kb;
  const float* Bp1 = Bp0 + 16 * 256;
  const float* Bp2 = Bp1 + 16 * 256;
  const float* Ap  = &h_lds[lr * 256 + kb];
  float bh0 = bhh[nt0 * 16 + lr];
  float bh1 = bhh[(nt0 + 1) * 16 + lr];
  float bh2 = bhh[(nt0 + 2) * 16 + lr];

  // phase-2 ownership: b = wave, j = (lane*8)..(lane*8+7)
  int b2 = wave;
  int j0 = (tid & 31) * 8;
  int nvb = lengths[b2] >> 1;

  for (int s = 0; s < 1024; ++s) {
    v8f c0 = {0,0,0,0,0,0,0,0}, c1 = c0, c2 = c0;
    for (int k = 0; k < 256; k += 4) {
      v2f a;  a.x  = Ap[k];  a.y  = Ap[k + 1];
      v2f b0; b0.x = Bp0[k]; b0.y = Bp0[k + 1];
      v2f b1; b1.x = Bp1[k]; b1.y = Bp1[k + 1];
      v2f b1b; b1b.x = Bp2[k]; b1b.y = Bp2[k + 1];
      c0 = __builtin_amdgcn_wmma_f32_16x16x4_f32(false, a, false, b0,  (short)0, c0, false, false);
      c1 = __builtin_amdgcn_wmma_f32_16x16x4_f32(false, a, false, b1,  (short)0, c1, false, false);
      c2 = __builtin_amdgcn_wmma_f32_16x16x4_f32(false, a, false, b1b, (short)0, c2, false, false);
    }
#pragma unroll
    for (int r = 0; r < 8; ++r) {
      int m = r + ((lane >> 4) << 3);  // batch row
      hp_lds[m * 768 + nt0 * 16 + lr]       = c0[r] + bh0;
      hp_lds[m * 768 + (nt0 + 1) * 16 + lr] = c1[r] + bh1;
      hp_lds[m * 768 + (nt0 + 2) * 16 + lr] = c2[r] + bh2;
    }
    __syncthreads();

    int tx = s;
    if (dir) { tx = nvb - 1 - s; if (tx < 0) tx = 0; }
    const float* xr = xp + (size_t)(tx * 16 + b2) * 768;
    float* op = out + (size_t)((b2 << 10) + s) * out_ld;
#pragma unroll
    for (int e = 0; e < 8; ++e) {
      int j = j0 + e;
      float x_r = xr[j], x_z = xr[256 + j], x_n = xr[512 + j];
      float h_r = hp_lds[b2 * 768 + j];
      float h_z = hp_lds[b2 * 768 + 256 + j];
      float h_n = hp_lds[b2 * 768 + 512 + j];
      float ho  = h_lds[b2 * 256 + j];
      float rg = 1.f / (1.f + expf(-(x_r + h_r)));
      float zg = 1.f / (1.f + expf(-(x_z + h_z)));
      float ng = tanhf(x_n + rg * h_n);
      float hn = (1.f - zg) * ng + zg * ho;
      op[j] = hn;
      h_lds[b2 * 256 + j] = hn;  // owner thread: no race
    }
    __syncthreads();
  }
}

// ---------------------------------------------------------------------------
// output head + ragged gather: d_out row r -> (b,t) with t < lengths[b]/2
// ---------------------------------------------------------------------------
__global__ void k_head(const float* __restrict__ h,     // (16,1024,512)
                       const float* __restrict__ fcw,   // (61,512)
                       const float* __restrict__ fcb,   // (61)
                       const int* __restrict__ lengths,
                       float* __restrict__ out, int nrows) {
  int row = blockIdx.x;
  if (row >= nrows) return;
  int acc = 0, b = 0, t = 0;
  for (int i = 0; i < 16; ++i) {
    int nf = lengths[i] >> 1;
    if (row < acc + nf) { b = i; t = row - acc; break; }
    acc += nf;
  }
  int o = threadIdx.x;
  if (o >= 61) return;
  const float* hr = h + (size_t)((b << 10) + t) * 512;
  const float* wr = fcw + o * 512;
  float s = fcb[o];
  for (int k = 0; k < 512; ++k) s += hr[k] * wr[k];
  out[(size_t)row * 61 + o] = s;
}

// ---------------------------------------------------------------------------
// host
// ---------------------------------------------------------------------------
static inline char* wsal(char** p, size_t bytes) {
  char* r = *p;
  *p += (bytes + 255) & ~(size_t)255;
  return r;
}

extern "C" void kernel_launch(void* const* d_in, const int* in_sizes, int n_in,
                              void* d_out, int out_size, void* d_ws, size_t ws_size,
                              hipStream_t stream) {
  (void)in_sizes; (void)n_in; (void)ws_size;
  const float* batch   = (const float*)d_in[0];
  const int*   lengths = (const int*)d_in[1];
  auto F = [&](int i) { return (const float*)d_in[i]; };
  // pytree (sorted-key) param order:
  const float *bn32_b=F(2),*bn32_g=F(3),*bn32_m=F(4),*bn32_v=F(5);
  const float *bn4_b =F(6),*bn4_g =F(7),*bn4_m =F(8),*bn4_v =F(9);
  const float *bn8_b =F(10),*bn8_g=F(11),*bn8_m=F(12),*bn8_v=F(13);
  const float *fc1_b=F(38),*fc1_w=F(39),*fc_b=F(40),*fc_w=F(41);
  const float *g0b_bhh=F(42),*g0b_bih=F(43),*g0b_whh=F(44),*g0b_wih=F(45);
  const float *g0f_bhh=F(46),*g0f_bih=F(47),*g0f_whh=F(48),*g0f_wih=F(49);
  const float *g1b_bhh=F(50),*g1b_bih=F(51),*g1b_whh=F(52),*g1b_wih=F(53);
  const float *g1f_bhh=F(54),*g1f_bih=F(55),*g1f_whh=F(56),*g1f_wih=F(57);

  // workspace carve
  char* p = (char*)d_ws;
  float* feats   = (float*)wsal(&p, 16 * 10740 * 4);
  bf16*  featsbf = (bf16*) wsal(&p, (size_t)16 * 10752 * 2);
  bf16*  w1bf    = (bf16*) wsal(&p, (size_t)1024 * 10752 * 2);
  float* rvec    = (float*)wsal(&p, 16 * 1024 * 4);
  float* p1      = (float*)wsal(&p, (size_t)16 * 8 * 1022 * 30 * 4);
  float* r2      = (float*)wsal(&p, (size_t)16 * 32 * 510 * 14 * 4);
  float* r3      = (float*)wsal(&p, (size_t)16 * 4 * 510 * 14 * 4);
  bf16*  Abuf    = (bf16*) wsal(&p, (size_t)16384 * 1152 * 2);
  bf16*  wbf_f   = (bf16*) wsal(&p, (size_t)768 * 1152 * 2);
  bf16*  wbf_b   = (bf16*) wsal(&p, (size_t)768 * 1152 * 2);
  float* xpf     = (float*)wsal(&p, (size_t)16384 * 768 * 4);
  float* xpb     = (float*)wsal(&p, (size_t)16384 * 768 * 4);
  float* h0      = (float*)wsal(&p, (size_t)16 * 1024 * 512 * 4);
  float* h1      = (float*)wsal(&p, (size_t)16 * 1024 * 512 * 4);
  float* ob      = (float*)wsal(&p, (size_t)16 * 1024 * 256 * 4);

  const int TB = 256;
  auto nb = [&](int n) { return (n + TB - 1) / TB; };

  // ---- CNN towers ----
  struct TW { int k1,s1,k2,s2,PH,PW,H2,W2,OH,OW,foff; const float *w1,*b1,*w2,*b2,*w3,*b3; };
  TW tw[4] = {
    { 4,1,4,2, 1022,30, 510,14, 140,14,     0, F(19),F(18),F(27),F(26),F(35),F(34)},
    { 8,2,3,2,  510,14, 254, 6,  70, 6,  7840, F(21),F(20),F(29),F(28),F(37),F(36)},
    {16,4,3,1,  254, 6, 252, 4,  68, 4,  9520, F(15),F(14),F(23),F(22),F(31),F(30)},
    {32,8,2,1,  126, 2, 125, 1,  33, 1, 10608, F(17),F(16),F(25),F(24),F(33),F(32)},
  };
  for (int i = 0; i < 4; ++i) {
    TW& t = tw[i];
    int n1 = 16 * 8 * t.PH * t.PW;
    k_conv1_pool<<<nb(n1), TB, 0, stream>>>(batch, t.w1, t.b1, bn8_g, bn8_b, bn8_m, bn8_v,
                                            p1, t.k1, t.s1, t.PH, t.PW);
    int n2 = 16 * 32 * t.H2 * t.W2;
    k_conv2<<<nb(n2), TB, 0, stream>>>(p1, t.w2, t.b2, bn32_g, bn32_b, bn32_m, bn32_v,
                                       r2, t.k2, t.s2, t.PH, t.PW, t.H2, t.W2);
    int n3 = 16 * 4 * t.H2 * t.W2;
    k_conv3<<<nb(n3), TB, 0, stream>>>(r2, t.w3, t.b3, bn4_g, bn4_b, bn4_m, bn4_v,
                                       r3, t.H2, t.W2);
    int n4 = 16 * 4 * t.OH * t.OW;
    k_resize<<<nb(n4), TB, 0, stream>>>(r3, lengths, feats, t.H2, t.W2, t.OH, t.OW, t.foff);
  }

  // ---- FC1 (bf16 WMMA, K padded 10740->10752) ----
  k_cvt_pad<<<nb(16 * 10752), TB, 0, stream>>>(feats, featsbf, 16, 10740, 10752);
  k_cvt_pad<<<nb(1024 * 10752), TB, 0, stream>>>(fc1_w, w1bf, 1024, 10740, 10752);
  {
    int tiles = (16 / 16) * (1024 / 64);
    k_gemm_bf16<<<(tiles + 7) / 8, TB, 0, stream>>>(featsbf, w1bf, fc1_b, rvec,
                                                    16, 1024, 10752, 1024 / 64);
  }

  // ---- GRU layer 0 input projections ----
  k_build_A0<<<nb(16384 * 1152), TB, 0, stream>>>(batch, rvec, Abuf);
  k_cvt_pad<<<nb(768 * 1152), TB, 0, stream>>>(g0f_wih, wbf_f, 768, 1152, 1152);
  k_cvt_pad<<<nb(768 * 1152), TB, 0, stream>>>(g0b_wih, wbf_b, 768, 1152, 1152);
  {
    int tiles = (16384 / 16) * (768 / 64);
    k_gemm_bf16<<<(tiles + 7) / 8, TB, 0, stream>>>(Abuf, wbf_f, g0f_bih, xpf,
                                                    16384, 768, 1152, 768 / 64);
    k_gemm_bf16<<<(tiles + 7) / 8, TB, 0, stream>>>(Abuf, wbf_b, g0b_bih, xpb,
                                                    16384, 768, 1152, 768 / 64);
  }
  // ---- layer 0 scans (f32 WMMA recurrence) ----
  k_gru_scan<<<2, 512, 0, stream>>>(xpf, xpb, g0f_whh, g0b_whh, g0f_bhh, g0b_bhh,
                                    lengths, h0, ob);
  k_scatter_back<<<nb(16 * 1024 * 256), TB, 0, stream>>>(ob, lengths, h0);

  // ---- GRU layer 1 ----
  k_build_A1<<<nb(16384 * 512), TB, 0, stream>>>(h0, Abuf);
  k_cvt_pad<<<nb(768 * 512), TB, 0, stream>>>(g1f_wih, wbf_f, 768, 512, 512);
  k_cvt_pad<<<nb(768 * 512), TB, 0, stream>>>(g1b_wih, wbf_b, 768, 512, 512);
  {
    int tiles = (16384 / 16) * (768 / 64);
    k_gemm_bf16<<<(tiles + 7) / 8, TB, 0, stream>>>(Abuf, wbf_f, g1f_bih, xpf,
                                                    16384, 768, 512, 768 / 64);
    k_gemm_bf16<<<(tiles + 7) / 8, TB, 0, stream>>>(Abuf, wbf_b, g1b_bih, xpb,
                                                    16384, 768, 512, 768 / 64);
  }
  k_gru_scan<<<2, 512, 0, stream>>>(xpf, xpb, g1f_whh, g1b_whh, g1f_bhh, g1b_bhh,
                                    lengths, h1, ob);
  k_scatter_back<<<nb(16 * 1024 * 256), TB, 0, stream>>>(ob, lengths, h1);

  // ---- head + ragged gather ----
  int nrows = out_size / 61;
  k_head<<<nrows, 64, 0, stream>>>(h1, fc_w, fc_b, lengths, (float*)d_out, nrows);
}
